// NeuroselectiveLinear_78932908966351
// MI455X (gfx1250) — compile-verified, matrix operand order checked by reading, not run on hardware
//
#include <hip/hip_runtime.h>

#define IN_F   4096
#define OUT_F  16384
#define ACT_IN 2048
#define ACT_OUT 8192
#define NROWS  8192   // B * S

#define TMA 256       // C-tile rows per workgroup
#define TNB 128       // C-tile cols per workgroup
#define TK  64        // staged K per buffer (2 x 32-K WMMA sub-steps)

typedef __attribute__((ext_vector_type(16))) __bf16    v16bf;
typedef __attribute__((ext_vector_type(8)))  float     v8f;
typedef __attribute__((ext_vector_type(4)))  unsigned  v4u;

union Frag { v16bf bf; v4u u[2]; };

static __device__ __forceinline__ unsigned short f2bf(float f) {
    unsigned u = __builtin_bit_cast(unsigned, f);
    unsigned r = u + 0x7FFFu + ((u >> 16) & 1u);   // round-to-nearest-even
    return (unsigned short)(r >> 16);
}

// Async global -> LDS copy (GVS form: SADDR base + 32-bit VGPR offset),
// 16B per lane, tracked by ASYNCcnt (CDNA5).
static __device__ __forceinline__ void async_b128(unsigned lds_off, unsigned voff,
                                                  unsigned long long sbase) {
    asm volatile("global_load_async_to_lds_b128 %0, %1, %2"
                 :: "v"(lds_off), "v"(voff), "s"(sbase)
                 : "memory");
}
static __device__ __forceinline__ void wait_async0() {
    asm volatile("s_wait_asynccnt 0x0" ::: "memory");
}
static __device__ __forceinline__ unsigned lds_off32(const void* p) {
    return (unsigned)(unsigned long long)p;   // low 32 bits = LDS offset
}

// ---------------------------------------------------------------- zero output
__global__ void zero_out_kernel(float4* __restrict__ out, long n4) {
    long i = (long)blockIdx.x * blockDim.x + threadIdx.x;
    long stride = (long)gridDim.x * blockDim.x;
    float4 z = make_float4(0.f, 0.f, 0.f, 0.f);
    for (; i < n4; i += stride) out[i] = z;
}

// ------------------------------------------------- gather x + convert to bf16
__global__ void gather_cvt_x_kernel(const float* __restrict__ x,
                                    const int*   __restrict__ in_idx,
                                    unsigned short* __restrict__ A) {
    long t = (long)blockIdx.x * blockDim.x + threadIdx.x;  // < NROWS*ACT_IN/2
    int n  = (int)(t >> 10);            // ACT_IN/2 == 1024
    int k  = ((int)t & 1023) * 2;
    int i0 = in_idx[k], i1 = in_idx[k + 1];
    float f0 = x[(long)n * IN_F + i0];
    float f1 = x[(long)n * IN_F + i1];
    ((unsigned*)A)[t] = (unsigned)f2bf(f0) | ((unsigned)f2bf(f1) << 16);
}

// -------------------------------------------------- convert weight to bf16
__global__ void cvt_w_kernel(const float2* __restrict__ w,
                             unsigned short* __restrict__ W) {
    long t = (long)blockIdx.x * blockDim.x + threadIdx.x;  // < ACT_OUT*ACT_IN/2
    float2 f = w[t];
    ((unsigned*)W)[t] = (unsigned)f2bf(f.x) | ((unsigned)f2bf(f.y) << 16);
}

// ------------------------------------------------------------- main GEMM
// C(8192x8192) = A(8192x2048) * W(8192x2048)^T, bf16 in / f32 accumulate.
// 256x128 C-tile per 8-wave WG, 64x64 per wave, double-buffered async LDS.
__launch_bounds__(256)
__global__ void nslinear_wmma_kernel(const unsigned short* __restrict__ A,
                                     const unsigned short* __restrict__ W,
                                     const float* __restrict__ bias,
                                     const int*   __restrict__ out_idx,
                                     float* __restrict__ out) {
    __shared__ unsigned short As[2][TMA * TK];   // 2 x 32 KB
    __shared__ unsigned short Bs[2][TNB * TK];   // 2 x 16 KB

    const int tid   = threadIdx.x;
    const int lane  = tid & 31;
    const int wave  = tid >> 5;
    const int waveM = wave >> 1;    // 0..3 -> 64-row strip
    const int waveN = wave & 1;     // 0..1 -> 64-col strip
    const int mBase = blockIdx.y * TMA;
    const int nBase = blockIdx.x * TNB;

    const int mrow    = lane & 15;
    const int halfSel = lane >> 4;          // 0 or 1
    const int kLo     = halfSel * 8;        // bf16 elems: K 0..7 | 8..15
    const int kHi     = kLo + 16;           // K 16..23 | 24..31

    // ---- staging geometry: rows of TK=64 bf16 = 128 B = 8 x 16B chunks
    const int sRow = tid >> 3;              // 0..31
    const int sCol = (tid & 7) * 16;        // byte offset within row

    // loop-invariant 32-bit global byte offsets (GVS VADDR)
    const unsigned voffA = (unsigned)((mBase + sRow) * (ACT_IN * 2)) + sCol;
    const unsigned voffB = (unsigned)((nBase + sRow) * (ACT_IN * 2)) + sCol;
    // loop-invariant LDS byte offsets (buffer 0)
    const unsigned ldsA  = lds_off32(&As[0][0]) + sRow * (TK * 2) + sCol;
    const unsigned ldsB  = lds_off32(&Bs[0][0]) + sRow * (TK * 2) + sCol;

    v8f acc[4][4];
#pragma unroll
    for (int mi = 0; mi < 4; ++mi)
#pragma unroll
        for (int ni = 0; ni < 4; ++ni) acc[mi][ni] = (v8f)0.0f;

    // ---- prologue: async-fill buffer 0 with K-tile 0
#pragma unroll
    for (int it = 0; it < 8; ++it)          // A: 8 chunks/thread (32-row step)
        async_b128(ldsA + it * (32 * TK * 2), voffA + it * (32 * ACT_IN * 2),
                   (unsigned long long)A);
#pragma unroll
    for (int it = 0; it < 4; ++it)          // B: 4 chunks/thread
        async_b128(ldsB + it * (32 * TK * 2), voffB + it * (32 * ACT_IN * 2),
                   (unsigned long long)W);
    wait_async0();
    __syncthreads();

    int cur = 0;
    for (int kt = 0; kt < ACT_IN; kt += TK, cur ^= 1) {
        // ---- issue async copies for next K-tile into the other buffer
        if (kt + TK < ACT_IN) {
            const int nxt = cur ^ 1;
            const unsigned ldsSelA = ldsA + nxt * (TMA * TK * 2);
            const unsigned ldsSelB = ldsB + nxt * (TNB * TK * 2);
            const unsigned long long sbA =
                (unsigned long long)A + (size_t)(kt + TK) * 2;
            const unsigned long long sbB =
                (unsigned long long)W + (size_t)(kt + TK) * 2;
#pragma unroll
            for (int it = 0; it < 8; ++it)
                async_b128(ldsSelA + it * (32 * TK * 2),
                           voffA + it * (32 * ACT_IN * 2), sbA);
#pragma unroll
            for (int it = 0; it < 4; ++it)
                async_b128(ldsSelB + it * (32 * TK * 2),
                           voffB + it * (32 * ACT_IN * 2), sbB);
        }

        // ---- two 32-K WMMA sub-steps over the staged 64-K tile
#pragma unroll
        for (int ks = 0; ks < TK; ks += 32) {
            Frag a[4], b[4];
#pragma unroll
            for (int mi = 0; mi < 4; ++mi) {
                int r = waveM * 64 + mi * 16 + mrow;
                a[mi].u[0] = *(const v4u*)&As[cur][r * TK + ks + kLo];
                a[mi].u[1] = *(const v4u*)&As[cur][r * TK + ks + kHi];
            }
#pragma unroll
            for (int ni = 0; ni < 4; ++ni) {
                int c = waveN * 64 + ni * 16 + mrow;
                b[ni].u[0] = *(const v4u*)&Bs[cur][c * TK + ks + kLo];
                b[ni].u[1] = *(const v4u*)&Bs[cur][c * TK + ks + kHi];
            }
#pragma unroll
            for (int mi = 0; mi < 4; ++mi)
#pragma unroll
                for (int ni = 0; ni < 4; ++ni)
                    acc[mi][ni] = __builtin_amdgcn_wmma_f32_16x16x32_bf16(
                        false, a[mi].bf, false, b[ni].bf,
                        (short)0, acc[mi][ni], false, false);
        }

        // ---- close the pipeline stage
        wait_async0();
        __syncthreads();
    }

    // -------- epilogue: bias + scatter columns through out_idx
#pragma unroll
    for (int ni = 0; ni < 4; ++ni) {
        int nAct = nBase + waveN * 64 + ni * 16 + mrow;  // active-col index
        int oc   = out_idx[nAct];
        float bv = bias[nAct];
#pragma unroll
        for (int mi = 0; mi < 4; ++mi) {
            int r0 = mBase + waveM * 64 + mi * 16 + halfSel * 8;
            v8f v = acc[mi][ni];
#pragma unroll
            for (int e = 0; e < 8; ++e)
                out[(size_t)(r0 + e) * OUT_F + oc] = v[e] + bv;
        }
    }
}

// ------------------------------------------------------------------ launcher
extern "C" void kernel_launch(void* const* d_in, const int* in_sizes, int n_in,
                              void* d_out, int out_size, void* d_ws, size_t ws_size,
                              hipStream_t stream) {
    const float* x      = (const float*)d_in[0];
    const float* weight = (const float*)d_in[1];
    const float* bias   = (const float*)d_in[2];
    const int*   in_idx = (const int*)d_in[3];
    const int*   out_idx= (const int*)d_in[4];
    float* out = (float*)d_out;

    unsigned short* Abf = (unsigned short*)d_ws;                       // 33.5 MB
    unsigned short* Wbf = Abf + (size_t)NROWS * ACT_IN;                // 33.5 MB

    // 1) zero the full output (inactive columns must be 0)
    long n4 = (long)NROWS * OUT_F / 4;
    zero_out_kernel<<<8192, 256, 0, stream>>>((float4*)out, n4);

    // 2) gather + convert x -> bf16 A, convert weight -> bf16 W
    long gx = (long)NROWS * (ACT_IN / 2);        // 8,388,608
    gather_cvt_x_kernel<<<(int)(gx / 256), 256, 0, stream>>>(x, in_idx, Abf);
    long gw = (long)ACT_OUT * (ACT_IN / 2);      // 8,388,608
    cvt_w_kernel<<<(int)(gw / 256), 256, 0, stream>>>((const float2*)weight, Wbf);

    // 3) WMMA GEMM + bias + column scatter
    dim3 grid(ACT_OUT / TNB, NROWS / TMA);       // 64 x 32
    nslinear_wmma_kernel<<<grid, 256, 0, stream>>>(Abf, Wbf, bias, out_idx, out);
}
